// DecorrelateBN_62689342652815
// MI455X (gfx1250) — matrix operand order repaired and checked by reading, not compiled, not run on hardware
//
#include <hip/hip_runtime.h>

// DecorrelateBN (IterNorm-style) for MI455X / gfx1250.
// B=32,N=4096,C=512,G=64 -> Ntot=131072 points, 8 channels/group.
//
// Roofline: ~17 GFLOP carried by v_wmma_f32_16x16x32_bf16 (free), traffic
// ~0.8GB (1 read of x for moments, 1 read + 1 write for apply) -> ~34us
// @ 23.3 TB/s. Means are fused into the moment pass via an augmented
// mask-row tile: M2[64,h] = channel-group sums, so only two passes over x.

#define C_TOT   512
#define G_GRP   64
#define NTOT    (32 * 4096)
#define EPSV    1e-4f
#define N_ITERS 5
#define NPART   64          // number of partial moment matrices
#define PSTRIDE 5184        // 80*64 moments + cnt + pad (floats)

typedef __attribute__((ext_vector_type(16))) __bf16 bf16x16;
typedef __attribute__((ext_vector_type(8)))  float  f32x8;

union BF16Frag { bf16x16 v; __bf16 e[16]; };

__device__ __forceinline__ f32x8 zero8() {
    f32x8 z;
#pragma unroll
    for (int i = 0; i < 8; ++i) z[i] = 0.0f;
    return z;
}

__device__ __forceinline__ f32x8 wmma_bf16(bf16x16 a, bf16x16 b, f32x8 c) {
    return __builtin_amdgcn_wmma_f32_16x16x32_bf16(
        /*neg_a=*/false, a, /*neg_b=*/false, b,
        /*c_mod=*/(short)0, c, /*reuse_a=*/false, /*reuse_b=*/false);
}

// Load 8 contiguous channels of one point, mask-zero, convert to bf16.
__device__ __forceinline__ void load8z(const float* p, float mk, __bf16* d) {
    const float4* q = reinterpret_cast<const float4*>(p);
    float4 a = q[0], b = q[1];
    d[0] = (__bf16)(a.x * mk);
    d[1] = (__bf16)(a.y * mk);
    d[2] = (__bf16)(a.z * mk);
    d[3] = (__bf16)(a.w * mk);
    d[4] = (__bf16)(b.x * mk);
    d[5] = (__bf16)(b.y * mk);
    d[6] = (__bf16)(b.z * mk);
    d[7] = (__bf16)(b.w * mk);
}

// ---------------------------------------------------------------------------
// K1: augmented second moments via WMMA bf16. 64 blocks x 8 waves.
// K-dim enumerated as j = n*8 + k so every fragment row is 8 contiguous
// channels of one point. A has 5 row-tiles: rows 0..63 = Xz (masked x),
// row 64 = mask vector (so tile 4 accumulates per-group channel sums S[h]).
// Each wave owns 256 points (64 steps of 4 points), keeps the full 80x64
// f32 accumulation in VGPRs, then stages wave-by-wave into LDS.
// ---------------------------------------------------------------------------
__global__ void __launch_bounds__(256)
k1_moments(const float* __restrict__ x, const unsigned char* __restrict__ mask,
           float* __restrict__ covPart) {
    const int t = threadIdx.x;
    const int lane = t & 31, wv = t >> 5;
    const int m = lane & 15, hi = lane >> 4;
    const int waveId = blockIdx.x * 8 + wv;   // 512 waves total
    const int base = waveId * 256;

    f32x8 acc[5][4];
#pragma unroll
    for (int a = 0; a < 5; ++a)
#pragma unroll
        for (int b = 0; b < 4; ++b) acc[a][b] = zero8();

    int cR[4];
#pragma unroll
    for (int tg = 0; tg < 4; ++tg) cR[tg] = 8 * (tg * 16 + m);

    float wcnt = 0.f;   // mask count (lane 0 of each wave)

    for (int s = 0; s < 64; ++s) {
        const int n0 = base + s * 4;
        // A operand points (16-bit A layout): low lanes K0-7 -> n0, K16-23 -> n2
        const int pa = n0 + hi, pb = n0 + 2 + hi;
        const float mka = mask[pa] ? 1.f : 0.f;
        const float mkb = mask[pb] ? 1.f : 0.f;
        const float* ra = x + (size_t)pa * C_TOT;
        const float* rb = x + (size_t)pb * C_TOT;

        BF16Frag afr[5];
#pragma unroll
        for (int tg = 0; tg < 4; ++tg) {
            load8z(ra + cR[tg], mka, &afr[tg].e[0]);
            load8z(rb + cR[tg], mkb, &afr[tg].e[8]);
        }
        // Mask-row tile: global rows 64..79, only row 64 (m==0) nonzero.
        {
            const __bf16 z  = (__bf16)0.f;
            const __bf16 ea = (m == 0) ? (__bf16)mka : z;
            const __bf16 eb = (m == 0) ? (__bf16)mkb : z;
#pragma unroll
            for (int i = 0; i < 8; ++i) { afr[4].e[i] = ea; afr[4].e[8 + i] = eb; }
        }
        // B operand points (B layout): low lanes K0-7 -> n0, K16-23 -> n2
        const int qa = n0 + hi, qb = qa + 2;
        const float mqa = mask[qa] ? 1.f : 0.f;
        const float mqb = mask[qb] ? 1.f : 0.f;
        const float* rqa = x + (size_t)qa * C_TOT;
        const float* rqb = x + (size_t)qb * C_TOT;

        if (lane == 0)  // points n0..n3 = mka, explicit n0+1, mkb, explicit n0+3
            wcnt += mka + (mask[n0 + 1] ? 1.f : 0.f) + mkb + (mask[n0 + 3] ? 1.f : 0.f);

#pragma unroll
        for (int th = 0; th < 4; ++th) {
            BF16Frag bfr;
            load8z(rqa + cR[th], mqa, &bfr.e[0]);
            load8z(rqb + cR[th], mqb, &bfr.e[8]);
#pragma unroll
            for (int tg = 0; tg < 5; ++tg)
                acc[tg][th] = wmma_bf16(afr[tg].v, bfr.v, acc[tg][th]);
        }
    }

    // Race-free wave-by-wave reduction into LDS.
    __shared__ float cv[80 * G_GRP];
    __shared__ float cntS;
    if (t == 0) cntS = 0.f;
    for (int idx = t; idx < 80 * G_GRP; idx += 256) cv[idx] = 0.f;
    __syncthreads();
    for (int w = 0; w < 8; ++w) {
        if (wv == w) {
#pragma unroll
            for (int tg = 0; tg < 5; ++tg)
#pragma unroll
                for (int th = 0; th < 4; ++th)
#pragma unroll
                    for (int i = 0; i < 8; ++i) {
                        const int row = tg * 16 + 8 * hi + i;
                        const int col = th * 16 + m;
                        cv[row * G_GRP + col] += acc[tg][th][i];
                    }
            if (lane == 0) cntS += wcnt;
        }
        __syncthreads();
    }
    for (int idx = t; idx < 80 * G_GRP; idx += 256)
        covPart[blockIdx.x * PSTRIDE + idx] = cv[idx];
    if (t == 0) covPart[blockIdx.x * PSTRIDE + 5120] = cntS;
}

// ---------------------------------------------------------------------------
// K2: reduce partials, build cov = M2/GN - mean*mean^T + eps*I, then
// Newton-Schulz inverse sqrt. 1 block, f32 in LDS (exactly 64KB: Y,Z,T,U;
// S/GN/means staged in U before the iterations need it).
// ---------------------------------------------------------------------------
__global__ void __launch_bounds__(256)
k2_ns(const float* __restrict__ covPart, float* __restrict__ meanOut,
      float* __restrict__ deconv) {
    const int t = threadIdx.x;
    __shared__ float Yb[4096], Zb[4096], Tb[4096], Ub[4096];

    // Stage S[h] (row 64 of moments) and GN into Ub.
    if (t < G_GRP) {
        float s = 0.f;
        for (int p = 0; p < NPART; ++p) s += covPart[p * PSTRIDE + 4096 + t];
        Ub[t] = s;
    }
    if (t == G_GRP) {
        float c = 0.f;
        for (int p = 0; p < NPART; ++p) c += covPart[p * PSTRIDE + 5120];
        Ub[G_GRP] = 8.0f * c;   // G_numel
    }
    __syncthreads();
    const float GN = Ub[G_GRP];
    const float invGN = 1.0f / GN;
    if (t < G_GRP) {
        const float mn = Ub[t] * invGN;   // per-group mean
        Ub[t] = mn;
        meanOut[t] = mn;
    }
    __syncthreads();

    // cov = M2/GN - mean_r*mean_c + eps*diag; Frobenius norm alongside.
    float sq = 0.f;
#pragma unroll
    for (int e = 0; e < 16; ++e) {
        const int idx = t + 256 * e;
        const int r = idx >> 6, c = idx & 63;
        float m2 = 0.f;
        for (int p = 0; p < NPART; ++p) m2 += covPart[p * PSTRIDE + idx];
        float a = m2 * invGN - Ub[r] * Ub[c] + ((r == c) ? EPSV : 0.f);
        Yb[idx] = a;
        sq += a * a;
    }
    Tb[t] = sq;
    __syncthreads();
    for (int s = 128; s > 0; s >>= 1) {
        if (t < s) Tb[t] += Tb[t + s];
        __syncthreads();
    }
    const float norm = sqrtf(Tb[0]);   // Frobenius norm
    __syncthreads();

    const float inv = 1.0f / norm;
#pragma unroll
    for (int e = 0; e < 16; ++e) {
        const int idx = t + 256 * e;
        const int r = idx >> 6, c = idx & 63;
        Yb[idx] *= inv;                 // Y = A / ||A||
        Zb[idx] = (r == c) ? 1.f : 0.f; // Z = I
    }
    __syncthreads();

    for (int it = 0; it < N_ITERS; ++it) {
        // T = 0.5*(3I - Z@Y)
#pragma unroll
        for (int e = 0; e < 16; ++e) {
            const int idx = t + 256 * e;
            const int r = idx >> 6, c = idx & 63;
            float d = 0.f;
            for (int h = 0; h < 64; ++h) d += Zb[r * 64 + h] * Yb[h * 64 + c];
            Tb[idx] = 0.5f * (((r == c) ? 3.f : 0.f) - d);
        }
        __syncthreads();
        // Y = Y@T
#pragma unroll
        for (int e = 0; e < 16; ++e) {
            const int idx = t + 256 * e;
            const int r = idx >> 6, c = idx & 63;
            float d = 0.f;
            for (int h = 0; h < 64; ++h) d += Yb[r * 64 + h] * Tb[h * 64 + c];
            Ub[idx] = d;
        }
        __syncthreads();
#pragma unroll
        for (int e = 0; e < 16; ++e) { const int idx = t + 256 * e; Yb[idx] = Ub[idx]; }
        __syncthreads();
        // Z = T@Z
#pragma unroll
        for (int e = 0; e < 16; ++e) {
            const int idx = t + 256 * e;
            const int r = idx >> 6, c = idx & 63;
            float d = 0.f;
            for (int h = 0; h < 64; ++h) d += Tb[r * 64 + h] * Zb[h * 64 + c];
            Ub[idx] = d;
        }
        __syncthreads();
#pragma unroll
        for (int e = 0; e < 16; ++e) { const int idx = t + 256 * e; Zb[idx] = Ub[idx]; }
        __syncthreads();
    }

    const float sc = rsqrtf(norm);      // deconv = Z / sqrt(||A||)
#pragma unroll
    for (int e = 0; e < 16; ++e) {
        const int idx = t + 256 * e;
        deconv[idx] = Zb[idx] * sc;
    }
}

// ---------------------------------------------------------------------------
// K3: apply whitening: out = (deconv @ Xc) * weight + bias, via WMMA.
// A = deconv fragments cached in VGPRs per wave; N-dim = (point,k) pairs,
// 2 points per step. 1024 blocks x 8 waves, 16 points per wave.
// global_prefetch_b8 pulls the next point-pair's rows while WMMA runs.
// ---------------------------------------------------------------------------
__global__ void __launch_bounds__(256)
k3_apply(const float* __restrict__ x, const unsigned char* __restrict__ mask,
         const float* __restrict__ meanG, const float* __restrict__ deconv,
         const float* __restrict__ weight, const float* __restrict__ bias,
         float* __restrict__ out) {
    __shared__ float sMean[G_GRP], sW[C_TOT], sB[C_TOT];
    const int t = threadIdx.x;
    if (t < G_GRP) sMean[t] = meanG[t];
    for (int c = t; c < C_TOT; c += 256) { sW[c] = weight[c]; sB[c] = bias[c]; }
    __syncthreads();

    const int lane = t & 31, wv = t >> 5;
    const int m = lane & 15, hi = lane >> 4;

    // Cache deconv A fragments: 4 g-tiles x 2 K-steps (h 0..31, 32..63)
    BF16Frag afr[4][2];
#pragma unroll
    for (int gt = 0; gt < 4; ++gt) {
        const int g = gt * 16 + m;
        const float* drow = deconv + g * 64;
#pragma unroll
        for (int ks = 0; ks < 2; ++ks) {
            const int h0 = ks * 32 + 8 * hi;
#pragma unroll
            for (int i = 0; i < 8; ++i) {
                afr[gt][ks].e[i]     = (__bf16)drow[h0 + i];
                afr[gt][ks].e[8 + i] = (__bf16)drow[h0 + 16 + i];
            }
        }
    }

    const int waveId = blockIdx.x * 8 + wv;   // 8192 waves
    const int base = waveId * 16;
    const int colp = m >> 3;                  // which of the 2 points is this lane's column
    const int k = m & 7;                      // channel-within-group

    for (int s = 0; s < 8; ++s) {
        const int p0 = base + s * 2;
        const int pp = p0 + colp;
        const float mk = mask[pp] ? 1.f : 0.f;
        const float* row = x + (size_t)pp * C_TOT;
        if (s < 7) __builtin_prefetch(row + 2 * C_TOT, 0, 1);

        f32x8 acc[4];
#pragma unroll
        for (int gt = 0; gt < 4; ++gt) acc[gt] = zero8();

#pragma unroll
        for (int ks = 0; ks < 2; ++ks) {
            BF16Frag bfr;
            const int hb = ks * 32 + 8 * hi;
#pragma unroll
            for (int i = 0; i < 16; ++i) {
                const int h = hb + i + (i >= 8 ? 8 : 0);
                bfr.e[i] = (__bf16)((row[8 * h + k] - sMean[h]) * mk);
            }
#pragma unroll
            for (int gt = 0; gt < 4; ++gt)
                acc[gt] = wmma_bf16(afr[gt][ks].v, bfr.v, acc[gt]);
        }

        float* orow = out + (size_t)pp * C_TOT;
#pragma unroll
        for (int gt = 0; gt < 4; ++gt)
#pragma unroll
            for (int i = 0; i < 8; ++i) {
                const int g = gt * 16 + 8 * hi + i;
                const int c = 8 * g + k;
                orow[c] = acc[gt][i] * sW[c] + sB[c];
            }
    }
}

// ---------------------------------------------------------------------------
extern "C" void kernel_launch(void* const* d_in, const int* in_sizes, int n_in,
                              void* d_out, int out_size, void* d_ws, size_t ws_size,
                              hipStream_t stream) {
    (void)in_sizes; (void)n_in; (void)out_size; (void)ws_size;
    // inputs: 0=coords (unused), 1=x, 2=mask(bool/u8), 3=weight, 4=bias
    const float*         x      = (const float*)d_in[1];
    const unsigned char* mask   = (const unsigned char*)d_in[2];
    const float*         weight = (const float*)d_in[3];
    const float*         bias   = (const float*)d_in[4];
    float* out = (float*)d_out;
    float* ws  = (float*)d_ws;

    float* covP  = ws;                        // 64 * 5184 = 331776 floats
    float* meanG = ws + (size_t)NPART * PSTRIDE;          // 64
    float* dec   = ws + (size_t)NPART * PSTRIDE + 64;     // 4096

    k1_moments<<<64,   256, 0, stream>>>(x, mask, covP);
    k2_ns     <<<1,    256, 0, stream>>>(covP, meanG, dec);
    k3_apply  <<<1024, 256, 0, stream>>>(x, mask, meanG, dec, weight, bias, out);
}
